// My_lstm_47425028882697
// MI455X (gfx1250) — matrix operand order repaired and checked by reading, not compiled, not run on hardware
//
#include <hip/hip_runtime.h>
#include <math.h>

#define B_   64
#define T_   128
#define IN_  1024
#define H_   2048
#define NG   8192              /* 4*H */
#define BT   8192              /* B*T */
#define BTH  (64u*128u*2048u)  /* elements per output tensor */

typedef __attribute__((ext_vector_type(16))) __bf16 bf16x16;
typedef __attribute__((ext_vector_type(8)))  float  f32x8;
typedef int v4i __attribute__((vector_size(16)));   // matches builtin pointee

#define GLOBAL_AS __attribute__((address_space(1)))
#define LDS_AS    __attribute__((address_space(3)))

#if __has_builtin(__builtin_amdgcn_global_load_async_to_lds_b128)
#define HAVE_ASYNC_LDS 1
#else
#define HAVE_ASYNC_LDS 0
#endif

#if __has_builtin(__builtin_amdgcn_tensor_load_to_lds) && __has_builtin(__builtin_amdgcn_s_wait_tensorcnt)
#define HAVE_TDM 1
#else
#define HAVE_TDM 0
#endif

#if HAVE_ASYNC_LDS
__device__ __forceinline__ void wait_async_le(int keep2) {
#if __has_builtin(__builtin_amdgcn_s_wait_asynccnt)
  if (keep2) __builtin_amdgcn_s_wait_asynccnt(2);
  else       __builtin_amdgcn_s_wait_asynccnt(0);
#else
  if (keep2) asm volatile("s_wait_asynccnt 2" ::: "memory");
  else       asm volatile("s_wait_asynccnt 0" ::: "memory");
#endif
}
#endif

#if HAVE_TDM
typedef unsigned int u32x4 __attribute__((ext_vector_type(4)));
typedef int          i32x4 __attribute__((ext_vector_type(4)));
typedef int          i32x8 __attribute__((ext_vector_type(8)));
#endif

__device__ __forceinline__ unsigned short f32_to_bf16_rne(float f) {
  union { float f; unsigned int u; } v; v.f = f;
  unsigned int u = v.u;
  u += 0x7fffu + ((u >> 16) & 1u);   // round-to-nearest-even
  return (unsigned short)(u >> 16);
}

// ---------------------------------------------------------------- convert ----
__global__ void k_convert(const float* __restrict__ src,
                          unsigned short* __restrict__ dst, int n) {
  int i = blockIdx.x * blockDim.x + threadIdx.x;
  int stride = gridDim.x * blockDim.x;
  for (; i < n; i += stride) dst[i] = f32_to_bf16_rne(src[i]);
}

// ------------------------------------------------- input-projection GEMM ----
// out[(2+g)*BTH + m*H + h] = sum_k x_bf[m,k]*Wx_bf[n,k] + bias[n]
// M = B*T = 8192, N = 4H = 8192, K = IN = 1024.  Block tile 128x128, 8 waves.
// Double-buffered LDS fed by async global->LDS DMA when available.
__global__ __launch_bounds__(256)
void k_gemm_x(const unsigned short* __restrict__ Abf,   // x_bf16   [8192 x 1024]
              const unsigned short* __restrict__ Wbf,   // Wx_bf16  [8192 x 1024]
              const float* __restrict__ bias,           // [8192]
              float* __restrict__ out) {
  const int K   = IN_;
  const int LDP = 40;                                 // padded LDS stride (halves)
  __shared__ __align__(16) unsigned short As[2][128 * 40];   // 2 x 10 KB
  __shared__ __align__(16) unsigned short Bs[2][128 * 40];   // 2 x 10 KB

  const int tid   = threadIdx.x;
  const int lane  = tid & 31;
  const int wave  = tid >> 5;
  const int l15   = lane & 15;
  const int lhalf = lane >> 4;              // 0 or 1
  const int wm    = wave >> 2;              // 0..1  (64-row slab)
  const int wn    = wave & 3;               // 0..3  (32-col slab)

  const int m0 = blockIdx.y * 128;
  const int n0 = blockIdx.x * 128;

  const int ldrow = tid >> 1;               // 0..127
  const int ldoff = (tid & 1) * 16;         // 0 or 16 halves

  f32x8 acc[4][2] = {};

#if HAVE_ASYNC_LDS
  // each thread DMAs one b128 of A and one b128 of B per tile
  auto issue = [&](int buf, int k0) {
    const unsigned short* gp = Abf + (size_t)(m0 + ldrow) * K + k0 + ldoff;
    const unsigned short* hp = Wbf + (size_t)(n0 + ldrow) * K + k0 + ldoff;
    __builtin_amdgcn_global_load_async_to_lds_b128(
        (GLOBAL_AS v4i*)gp, (LDS_AS v4i*)&As[buf][ldrow * LDP + ldoff], 0, 0);
    __builtin_amdgcn_global_load_async_to_lds_b128(
        (GLOBAL_AS v4i*)hp, (LDS_AS v4i*)&Bs[buf][ldrow * LDP + ldoff], 0, 0);
  };
  const int nk = K / 32;
  issue(0, 0);
  for (int i = 0; i < nk; ++i) {
    const int cur  = i & 1;
    const int more = (i + 1 < nk);
    if (more) issue(cur ^ 1, (i + 1) * 32);   // overlap DMA with compute
    wait_async_le(more);                      // newest pair may stay in flight
    __syncthreads();
#else
  for (int k0 = 0; k0 < K; k0 += 32) {
    const int cur = 0;
    const uint4 ga = *(const uint4*)(Abf + (size_t)(m0 + ldrow) * K + k0 + ldoff);
    const uint4 gb = *(const uint4*)(Wbf + (size_t)(n0 + ldrow) * K + k0 + ldoff);
    *(uint4*)(&As[0][ldrow * LDP + ldoff]) = ga;
    *(uint4*)(&Bs[0][ldrow * LDP + ldoff]) = gb;
    if (k0 + 32 < K) {
      __builtin_prefetch(Abf + (size_t)(m0 + ldrow) * K + k0 + 32 + ldoff, 0, 1);
      __builtin_prefetch(Wbf + (size_t)(n0 + ldrow) * K + k0 + 32 + ldoff, 0, 1);
    }
    __syncthreads();
#endif

    bf16x16 afr[4], bfr[2];
#pragma unroll
    for (int mt = 0; mt < 4; ++mt) {
      const unsigned short* p = &As[cur][(wm * 64 + mt * 16 + l15) * LDP + lhalf * 8];
      ((uint4*)&afr[mt])[0] = *(const uint4*)p;
      ((uint4*)&afr[mt])[1] = *(const uint4*)(p + 16);
    }
#pragma unroll
    for (int nt = 0; nt < 2; ++nt) {
      const unsigned short* p = &Bs[cur][(wn * 32 + nt * 16 + l15) * LDP + lhalf * 8];
      ((uint4*)&bfr[nt])[0] = *(const uint4*)p;
      ((uint4*)&bfr[nt])[1] = *(const uint4*)(p + 16);
    }
#pragma unroll
    for (int mt = 0; mt < 4; ++mt)
#pragma unroll
      for (int nt = 0; nt < 2; ++nt)
        acc[mt][nt] = __builtin_amdgcn_wmma_f32_16x16x32_bf16(
            false, afr[mt], false, bfr[nt], (short)0, acc[mt][nt], false, false);
    __syncthreads();
  }

  // scatter to the 4 gate output sections (+bias)
#pragma unroll
  for (int nt = 0; nt < 2; ++nt) {
    const int n  = n0 + wn * 32 + nt * 16 + l15;
    const int g  = n >> 11;
    const int h  = n & (H_ - 1);
    const float bv = bias[n];
    float* dst = out + (size_t)(2 + g) * BTH + h;
#pragma unroll
    for (int mt = 0; mt < 4; ++mt) {
      const int mbase = m0 + wm * 64 + mt * 16 + lhalf * 8;
#pragma unroll
      for (int v = 0; v < 8; ++v)
        dst[(size_t)(mbase + v) * H_] = acc[mt][nt][v] + bv;
    }
  }
}

// --------------------------------------------------- recurrent GEMM step ----
// out[(2+g)*BTH + (b*T+t)*H + h] += sum_k a_prev_bf[b,k] * Wa_bf[n,k]
// M = 64, N = 8192, K = 2048.  64 blocks of 128 cols; 8 waves x 16 cols each.
// a_prev tile is staged by the Tensor Data Mover when available; the D#
// pad fields reproduce the 40-half padded LDS stride in hardware.
__global__ __launch_bounds__(256)
void k_gemm_rec(const unsigned short* __restrict__ Abf,  // a_prev bf16 [64 x 2048]
                const unsigned short* __restrict__ Wbf,  // Wa_bf16 [8192 x 2048]
                float* __restrict__ out, int t) {
  const int K   = H_;
  const int LDP = 40;
  __shared__ __align__(16) unsigned short As[64 * 40];    // 5 KB

  const int tid   = threadIdx.x;
  const int lane  = tid & 31;
  const int wave  = tid >> 5;               // 0..7 -> 16-col slab
  const int l15   = lane & 15;
  const int lhalf = lane >> 4;

  const int n0 = blockIdx.x * 128;
  const int n  = n0 + wave * 16 + l15;

  f32x8 acc[4] = {};

  for (int k0 = 0; k0 < K; k0 += 32) {
#if HAVE_TDM
    if (wave == 0) {
      // D# for a 64x32(half) tile of the 64x2048 a_prev tensor, 2B elements,
      // padded in LDS: 16 DWORDs data + 4 DWORDs pad -> 40-half row stride.
      const unsigned long long ga =
          (unsigned long long)(uintptr_t)(Abf + k0);        // tile start
      const unsigned lds_base = (unsigned)(uintptr_t)As;    // generic low32 = LDS offset
      u32x4 g0;
      g0[0] = 1u;                                           // count=1, valid
      g0[1] = lds_base;                                     // lds_addr (bytes)
      g0[2] = (unsigned)ga;                                 // global_addr[31:0]
      g0[3] = (unsigned)(ga >> 32) | (2u << 30);            // addr[56:32] | type=2
      i32x8 g1;
      g1[0] = (int)((1u << 16)      // data_size = 2 bytes
                  | (1u << 20)      // pad_enable
                  | (3u << 22)      // pad_interval: 16 DWORDs
                  | (3u << 25));    // pad_amount:   4 DWORDs
      g1[1] = (int)((unsigned)H_ << 16);   // tensor_dim0 = 2048 (bits 79:48)
      g1[2] = (int)(64u << 16);            // tensor_dim1 = 64   (bits 111:80)
      g1[3] = (int)(32u << 16);            // tile_dim0 = 32     (bits 127:112)
      g1[4] = 64;                          // tile_dim1 = 64     (bits 143:128)
      g1[5] = H_;                          // tensor_dim0_stride (bits 207:160)
      g1[6] = 0;
      g1[7] = 0;
      i32x4 g2 = {0, 0, 0, 0};
      i32x4 g3 = {0, 0, 0, 0};
#if __clang_major__ >= 23
      i32x8 g23 = {0, 0, 0, 0, 0, 0, 0, 0};
      __builtin_amdgcn_tensor_load_to_lds(g0, g1, g2, g3, g23, 0);
#else
      __builtin_amdgcn_tensor_load_to_lds(g0, g1, g2, g3, 0);
#endif
      __builtin_amdgcn_s_wait_tensorcnt(0);
    }
    __syncthreads();
#else
    if (tid < 128) {                         // stage 64x32 a_prev tile
      const int row = tid >> 1, off = (tid & 1) * 16;
      *(uint4*)(&As[row * LDP + off]) =
          *(const uint4*)(Abf + (size_t)row * K + k0 + off);
    }
    __syncthreads();
#endif

    bf16x16 bfr;
    const unsigned short* pb = Wbf + (size_t)n * K + k0 + lhalf * 8;
    ((uint4*)&bfr)[0] = *(const uint4*)pb;
    ((uint4*)&bfr)[1] = *(const uint4*)(pb + 16);

#pragma unroll
    for (int mt = 0; mt < 4; ++mt) {
      bf16x16 afr;
      const unsigned short* pa = &As[(mt * 16 + l15) * LDP + lhalf * 8];
      ((uint4*)&afr)[0] = *(const uint4*)pa;
      ((uint4*)&afr)[1] = *(const uint4*)(pa + 16);
      acc[mt] = __builtin_amdgcn_wmma_f32_16x16x32_bf16(
          false, afr, false, bfr, (short)0, acc[mt], false, false);
    }
    __syncthreads();
  }

  const int g = n >> 11;
  const int h = n & (H_ - 1);
  float* dst = out + (size_t)(2 + g) * BTH + h;
#pragma unroll
  for (int mt = 0; mt < 4; ++mt) {
#pragma unroll
    for (int v = 0; v < 8; ++v) {
      const int b = mt * 16 + lhalf * 8 + v;
      dst[(size_t)(b * T_ + t) * H_] += acc[mt][v];
    }
  }
}

// ------------------------------------------------------- cell pointwise -----
__global__ void k_pointwise(float* __restrict__ out, const float* __restrict__ c0,
                            unsigned short* __restrict__ a_bf, int t) {
  int i = blockIdx.x * blockDim.x + threadIdx.x;   // over B*H
  if (i >= B_ * H_) return;
  const int b = i >> 11;            // H = 2048
  const int h = i & (H_ - 1);
  const size_t row = (size_t)(b * T_ + t) * H_ + h;

  const float yi = out[2u * BTH + row];
  const float yf = out[3u * BTH + row];
  const float yg = out[4u * BTH + row];
  const float yo = out[5u * BTH + row];
  const float cp = (t == 0) ? c0[i] : out[1u * BTH + row - H_];

  const float si = 1.f / (1.f + __expf(-yi));
  const float sf = 1.f / (1.f + __expf(-yf));
  const float so = 1.f / (1.f + __expf(-yo));
  const float c  = sf * cp + si * tanhf(yg);
  const float a  = so * tanhf(c);

  out[0u * BTH + row] = a;
  out[1u * BTH + row] = c;
  a_bf[i] = f32_to_bf16_rne(a);
}

// -------------------------------------------------------------- launcher ----
extern "C" void kernel_launch(void* const* d_in, const int* in_sizes, int n_in,
                              void* d_out, int out_size, void* d_ws, size_t ws_size,
                              hipStream_t stream) {
  (void)in_sizes; (void)n_in; (void)out_size; (void)ws_size;
  const float* x    = (const float*)d_in[0];   // [B,T,IN]
  const float* Wx   = (const float*)d_in[1];   // [4H,IN]
  const float* Wa   = (const float*)d_in[2];   // [4H,H]
  const float* bias = (const float*)d_in[3];   // [4H]
  const float* a0   = (const float*)d_in[4];   // [B,H]
  const float* c0   = (const float*)d_in[5];   // [B,H]
  float* out = (float*)d_out;

  unsigned short* ws    = (unsigned short*)d_ws;
  unsigned short* Wx_bf = ws;                                  // 8192*1024
  unsigned short* Wa_bf = Wx_bf + (size_t)NG * IN_;            // 8192*2048
  unsigned short* x_bf  = Wa_bf + (size_t)NG * H_;             // 8192*1024
  unsigned short* a_bf  = x_bf + (size_t)BT * IN_;             // 64*2048

  k_convert<<<2048, 256, 0, stream>>>(Wx, Wx_bf, NG * IN_);
  k_convert<<<4096, 256, 0, stream>>>(Wa, Wa_bf, NG * H_);
  k_convert<<<2048, 256, 0, stream>>>(x,  x_bf,  BT * IN_);
  k_convert<<<64,   256, 0, stream>>>(a0, a_bf,  B_ * H_);

  // time-independent projection: fills yi/yf/yg/yo sections with x@Wx^T + b
  k_gemm_x<<<dim3(64, 64), 256, 0, stream>>>(x_bf, Wx_bf, bias, out);

  // sequential recurrence
  for (int t = 0; t < T_; ++t) {
    k_gemm_rec<<<64, 256, 0, stream>>>(a_bf, Wa_bf, out, t);
    k_pointwise<<<(B_ * H_ + 255) / 256, 256, 0, stream>>>(out, c0, a_bf, t);
  }
}
